// DUPLEX_gat_32684701123297
// MI455X (gfx1250) — matrix operand order repaired and checked by reading, not compiled, not run on hardware
//
#include <hip/hip_runtime.h>

#define NNODES 50000
#define NEDGES 800000

typedef _Float16 fp16_t;
typedef __attribute__((ext_vector_type(8)))  _Float16 v8h;
typedef __attribute__((ext_vector_type(16))) _Float16 v16h;
typedef __attribute__((ext_vector_type(8)))  float    v8f;

// ---------- helpers: order-preserving float<->uint for atomicMax ----------
__device__ __forceinline__ unsigned f2ord(float f) {
  unsigned u = __float_as_uint(f);
  return (u & 0x80000000u) ? ~u : (u | 0x80000000u);
}
__device__ __forceinline__ float ord2f(unsigned u) {
  return __uint_as_float((u & 0x80000000u) ? (u ^ 0x80000000u) : ~u);
}

// ---------- utility kernels ----------
__global__ void k_zero(float* __restrict__ p, int n) {
  int i = blockIdx.x * blockDim.x + threadIdx.x;
  if (i < n) p[i] = 0.0f;
}

__global__ void k_f32_to_f16(const float* __restrict__ in, fp16_t* __restrict__ out, int n) {
  int i = blockIdx.x * blockDim.x + threadIdx.x;
  if (i < n) out[i] = (fp16_t)in[i];
}

// Pack 128x128 row-major W (K x N) into WMMA B-fragment order:
// Wp[((kb*2+hi)*128 + n)*16 + j] = W[(kb*32 + hi*16 + j)*128 + n]
__global__ void k_pack_w(const float* __restrict__ W, fp16_t* __restrict__ Wp) {
  int idx = blockIdx.x * blockDim.x + threadIdx.x;  // 16384 total
  int j  = idx & 15;
  int n  = (idx >> 4) & 127;
  int g  = idx >> 11;          // g = kb*2 + hi, 0..7
  int kb = g >> 1, hi = g & 1;
  int k  = kb * 32 + hi * 16 + j;
  Wp[idx] = (fp16_t)W[k * 128 + n];
}

// ---------- WMMA GEMM: C[M x 128] = A[M x 128](f16) * W[128 x 128](packed f16) ----------
// One wave per 16x16 output tile; 8 waves/block cover all 128 columns of a 16-row stripe.
__global__ void __launch_bounds__(256)
k_gemm_wmma(const fp16_t* __restrict__ A, const fp16_t* __restrict__ Wp, float* __restrict__ C) {
  const int wave = threadIdx.x >> 5;
  const int lane = threadIdx.x & 31;
  const int r  = lane & 15;
  const int hi = lane >> 4;
  const int m0 = blockIdx.x << 4;   // M = 50000 = 3125 * 16, no tail -> EXEC all ones
  const int n0 = wave << 4;

  const fp16_t* arow = A + (size_t)(m0 + r) * 128;
  v8f c = {};
#pragma unroll
  for (int kb = 0; kb < 4; ++kb) {
    const int k0 = kb * 32;
    // A fragment (contiguous K, matching packed B): lanes 0-15 hold K=[0..15], lanes 16-31 K=[16..31]
    v8h alo = *(const v8h*)(arow + k0 + hi * 16);
    v8h ahi = *(const v8h*)(arow + k0 + hi * 16 + 8);
    v16h a = __builtin_shufflevector(alo, ahi, 0,1,2,3,4,5,6,7,8,9,10,11,12,13,14,15);
    // B fragment: prepacked contiguous 16 halfs per lane
    v16h b = *(const v16h*)(Wp + ((size_t)((kb * 2 + hi) * 128 + (n0 + r))) * 16);
    c = __builtin_amdgcn_wmma_f32_16x16x32_f16(false, a, false, b, (short)0, c, false, false);
  }
  // C fragment: VGPR v -> row m0 + hi*8 + v, col n0 + r
  float* crow = C + (size_t)(m0 + hi * 8) * 128 + n0 + r;
#pragma unroll
  for (int v = 0; v < 8; ++v) crow[(size_t)v * 128] = c[v];
}

// ---------- attention scores: el[n,h] = <ft[n,h,:], al[h,:]>, er likewise ----------
__global__ void k_attn(const float* __restrict__ ft, const float* __restrict__ al,
                       const float* __restrict__ ar, float* __restrict__ el,
                       float* __restrict__ er, int n4) {
  int idx = blockIdx.x * blockDim.x + threadIdx.x;
  if (idx >= n4) return;
  int hh = idx & 3, n = idx >> 2;
  const float* f = ft + (size_t)n * 128 + hh * 32;
  float sl = 0.f, sr = 0.f;
#pragma unroll
  for (int i = 0; i < 32; ++i) { float v = f[i]; sl += v * al[hh * 32 + i]; sr += v * ar[hh * 32 + i]; }
  el[idx] = sl;
  er[idx] = sr;
}

// ---------- edge pass 1: e = leaky_relu(el[src]+er[dst]); segment max over dst ----------
__global__ void k_edge1(const int* __restrict__ src, const int* __restrict__ dst,
                        const float* __restrict__ el, const float* __restrict__ er,
                        float* __restrict__ eedge, unsigned* __restrict__ emaxu, int n) {
  int idx = blockIdx.x * blockDim.x + threadIdx.x;
  if (idx >= n) return;
  int hh = idx & 3, e = idx >> 2;
  int s = src[e], d = dst[e];
  float x = el[s * 4 + hh] + er[d * 4 + hh];
  x = (x > 0.f) ? x : 0.2f * x;      // NEG_SLOPE = 0.2
  eedge[idx] = x;
  atomicMax(&emaxu[d * 4 + hh], f2ord(x));
}

// ---------- edge pass 2: ex = exp(e - emax[dst]); segment sum ----------
__global__ void k_edge2(const int* __restrict__ dst, float* __restrict__ eedge,
                        const unsigned* __restrict__ emaxu, float* __restrict__ denom, int n) {
  int idx = blockIdx.x * blockDim.x + threadIdx.x;
  if (idx >= n) return;
  int hh = idx & 3, e = idx >> 2;
  int d = dst[e];
  float m  = ord2f(emaxu[d * 4 + hh]);
  float ex = __expf(eedge[idx] - m);
  eedge[idx] = ex;
  atomicAdd(&denom[d * 4 + hh], ex);
}

// ---------- edge pass 3: acc[dst] += (ex/denom)*w * ft[src]; one wave32 per edge ----------
__global__ void __launch_bounds__(256)
k_edge3(const int* __restrict__ src, const int* __restrict__ dst,
        const float* __restrict__ ex, const float* __restrict__ denom,
        const float* __restrict__ ew, const float* __restrict__ ft,
        float* __restrict__ acc, int nE) {
  int gid = blockIdx.x * blockDim.x + threadIdx.x;
  int e = gid >> 5;
  if (e >= nE) return;
  int lane = gid & 31;
  int s = src[e], d = dst[e];
  int hh = lane >> 3;                                   // 4 lanes-of-8 -> 4 heads
  float den = denom[d * 4 + hh];
  float a = ex[e * 4 + hh] / fmaxf(den, 1e-9f) * ew[e];
  const float4 v = *(const float4*)(ft + (size_t)s * 128 + lane * 4);
  float* o = acc + (size_t)d * 128 + lane * 4;
  atomicAdd(o + 0, a * v.x);
  atomicAdd(o + 1, a * v.y);
  atomicAdd(o + 2, a * v.z);
  atomicAdd(o + 3, a * v.w);
}

// ---------- layer-0 epilogue: h = relu(acc + b) -> fp16 for next layer's WMMA ----------
__global__ void k_epi0(const float* __restrict__ acc, const float* __restrict__ b,
                       fp16_t* __restrict__ hout, int n) {
  int i = blockIdx.x * blockDim.x + threadIdx.x;
  if (i >= n) return;
  int c = i & 127;
  float v = acc[i] + b[c];
  hout[i] = (fp16_t)fmaxf(v, 0.f);
}

// ---------- layer-1 epilogue: out = mean over heads of (acc + b) ----------
__global__ void k_epi1(const float* __restrict__ acc, const float* __restrict__ b,
                       float* __restrict__ o, int n) {
  int i = blockIdx.x * blockDim.x + threadIdx.x;
  if (i >= n) return;
  int f = i & 31, nn = i >> 5;
  float s = 0.f;
#pragma unroll
  for (int h = 0; h < 4; ++h) s += acc[(size_t)nn * 128 + h * 32 + f] + b[h * 32 + f];
  o[i] = 0.25f * s;
}

extern "C" void kernel_launch(void* const* d_in, const int* in_sizes, int n_in,
                              void* d_out, int out_size, void* d_ws, size_t ws_size,
                              hipStream_t stream) {
  (void)in_sizes; (void)n_in; (void)out_size; (void)ws_size;
  const int N = NNODES, E = NEDGES;

  const float* x_am     = (const float*)d_in[0];
  const float* x_ph     = (const float*)d_in[1];
  const float* exist    = (const float*)d_in[2];
  const float* am_exist = (const float*)d_in[3];
  const int*   src      = (const int*)d_in[4];
  const int*   dst      = (const int*)d_in[5];
  const float *W0a = (const float*)d_in[6],  *al0a = (const float*)d_in[7],
              *ar0a = (const float*)d_in[8], *b0a = (const float*)d_in[9];
  const float *W0p = (const float*)d_in[10], *al0p = (const float*)d_in[11],
              *ar0p = (const float*)d_in[12], *b0p = (const float*)d_in[13];
  const float *W1a = (const float*)d_in[14], *al1a = (const float*)d_in[15],
              *ar1a = (const float*)d_in[16], *b1a = (const float*)d_in[17];
  const float *W1p = (const float*)d_in[18], *al1p = (const float*)d_in[19],
              *ar1p = (const float*)d_in[20], *b1p = (const float*)d_in[21];
  float* out = (float*)d_out;

  // ---- carve workspace (~92 MB) ----
  size_t off = 0;
  auto carve = [&](size_t bytes) -> void* {
    void* p = (char*)d_ws + off;
    off += (bytes + 255) & ~(size_t)255;
    return p;
  };
  fp16_t*  xa16  = (fp16_t*)carve((size_t)N * 128 * 2);   // also reused as h_am (fp16)
  fp16_t*  xp16  = (fp16_t*)carve((size_t)N * 128 * 2);   // also reused as h_ph (fp16)
  fp16_t*  wp    = (fp16_t*)carve(128 * 128 * 2);
  float*   ft    = (float*)carve((size_t)N * 128 * 4);
  float*   el    = (float*)carve((size_t)N * 4 * 4);
  float*   er    = (float*)carve((size_t)N * 4 * 4);
  unsigned* emaxu = (unsigned*)carve((size_t)N * 4 * 4);
  float*   denom = (float*)carve((size_t)N * 4 * 4);
  float*   eedge = (float*)carve((size_t)E * 4 * 4);
  float*   acc   = (float*)carve((size_t)N * 128 * 4);

  auto cdiv = [](int a, int b) { return (a + b - 1) / b; };
  const int NF = N * 128;       // 6.4M
  const int N4 = N * 4;
  const int E4 = E * 4;

  // convert both input feature sets to fp16 once
  k_f32_to_f16<<<cdiv(NF, 256), 256, 0, stream>>>(x_am, xa16, NF);
  k_f32_to_f16<<<cdiv(NF, 256), 256, 0, stream>>>(x_ph, xp16, NF);

  auto run_conv = [&](const fp16_t* A16, const float* W, const float* al,
                      const float* ar, const float* ew) {
    k_pack_w<<<64, 256, 0, stream>>>(W, wp);
    k_gemm_wmma<<<N / 16, 256, 0, stream>>>(A16, wp, ft);
    k_attn<<<cdiv(N4, 256), 256, 0, stream>>>(ft, al, ar, el, er, N4);
    k_zero<<<cdiv(NF, 256), 256, 0, stream>>>(acc, NF);
    k_zero<<<cdiv(N4, 256), 256, 0, stream>>>((float*)emaxu, N4);   // ordered-uint 0 == -inf
    k_zero<<<cdiv(N4, 256), 256, 0, stream>>>(denom, N4);
    k_edge1<<<cdiv(E4, 256), 256, 0, stream>>>(src, dst, el, er, eedge, emaxu, E4);
    k_edge2<<<cdiv(E4, 256), 256, 0, stream>>>(dst, eedge, emaxu, denom, E4);
    k_edge3<<<E / 8, 256, 0, stream>>>(src, dst, eedge, denom, ew, ft, acc, E);
  };

  // layer 0, am stream -> h_am (fp16, overwrites xa16; stream-ordered so safe)
  run_conv(xa16, W0a, al0a, ar0a, am_exist);
  k_epi0<<<cdiv(NF, 256), 256, 0, stream>>>(acc, b0a, xa16, NF);
  // layer 0, ph stream
  run_conv(xp16, W0p, al0p, ar0p, exist);
  k_epi0<<<cdiv(NF, 256), 256, 0, stream>>>(acc, b0p, xp16, NF);
  // layer 1, am stream -> mean over heads -> d_out[0 : N*32)
  run_conv(xa16, W1a, al1a, ar1a, am_exist);
  k_epi1<<<cdiv(N * 32, 256), 256, 0, stream>>>(acc, b1a, out, N * 32);
  // layer 1, ph stream -> d_out[N*32 : 2*N*32)
  run_conv(xp16, W1p, al1p, ar1p, exist);
  k_epi1<<<cdiv(N * 32, 256), 256, 0, stream>>>(acc, b1p, out + (size_t)N * 32, N * 32);
}
